// MultiScaleRetention_23613730193586
// MI455X (gfx1250) — matrix operand order repaired and checked by reading, not compiled, not run on hardware
//
#include <hip/hip_runtime.h>
#include <hip/hip_bf16.h>
#include <cstdint>
#include <cstddef>

#define B_     2
#define S_     2048
#define H_     1024
#define HEADS_ 16
#define KD_    64
#define VDIM_  2048
#define HD_    128

typedef _Float16 f16;
typedef _Float16 v16h __attribute__((ext_vector_type(16)));
typedef _Float16 v8h  __attribute__((ext_vector_type(8)));
typedef float    v8f  __attribute__((ext_vector_type(8)));

__device__ __forceinline__ v16h cat16(v8h lo, v8h hi) {
  v16h r;
#pragma unroll
  for (int i = 0; i < 8; ++i) { r[i] = lo[i]; r[i + 8] = hi[i]; }
  return r;
}

__device__ __forceinline__ v8f wmma_f16(v16h a, v16h b, v8f c) {
  // D(16x16 f32) = A(16x32 f16) * B(32x16 f16) + C
  return __builtin_amdgcn_wmma_f32_16x16x32_f16(false, a, false, b, (short)0, c,
                                                false, false);
}

// ---------------------------------------------------------------------------
// f32 -> f16 conversion
// ---------------------------------------------------------------------------
__global__ void cvt_f32_f16(const float* __restrict__ src, f16* __restrict__ dst, int n) {
  int i = blockIdx.x * blockDim.x + threadIdx.x;
  if (i < n) dst[i] = (f16)src[i];
}

// ---------------------------------------------------------------------------
// Generic tiled WMMA GEMM:  C[M][N] = A[M][K] * Bm[N][K]^T   (Bm K-contiguous)
// Block tile 128x128, K-step 32, 8 waves (32x64 each), double-buffered LDS.
// Pipeline: [barrier] -> next-tile global loads -> regs | frag ds_loads |
//           WMMAs | deferred ds_stores of next tile.
// ---------------------------------------------------------------------------
enum { MODE_ROT = 0, MODE_VT = 1, MODE_F32 = 2, MODE_F16 = 3 };

#define BLK_M 128
#define BLK_N 128
#define BLK_K 32
#define LDS_STRIDE 40  // halfs; 80B row pitch (16B aligned), dodges bank conflicts

template <int MODE>
__global__ __launch_bounds__(256) void gemm_wmma(
    const f16* __restrict__ A, const f16* __restrict__ Bm, void* __restrict__ outp,
    int K, int Ncols, const float* __restrict__ sinp, const float* __restrict__ cosp,
    float scale) {
  __shared__ __align__(16) f16 sA[2][BLK_M * LDS_STRIDE];
  __shared__ __align__(16) f16 sB[2][BLK_N * LDS_STRIDE];

  const int tid  = threadIdx.x;
  const int wave = tid >> 5, lane = tid & 31;
  const int hh = lane >> 4, ln = lane & 15;
  const int wm = wave >> 1;  // 0..3 -> 32-row strip
  const int wn = wave & 1;   // 0..1 -> 64-col strip
  const int m0 = blockIdx.x * BLK_M;
  const int n0 = blockIdx.y * BLK_N;

  // per-thread staging coordinates (2 chunks each of A and B)
  const int srow0 = tid >> 2, sch = tid & 3;  // + rep*64 rows

  // prologue: stage tile 0 directly
#pragma unroll
  for (int rep = 0; rep < 2; ++rep) {
    int row = srow0 + rep * 64;
    v8h va = *(const v8h*)(A + (size_t)(m0 + row) * K + sch * 8);
    v8h vb = *(const v8h*)(Bm + (size_t)(n0 + row) * K + sch * 8);
    *(v8h*)(sA[0] + row * LDS_STRIDE + sch * 8) = va;
    *(v8h*)(sB[0] + row * LDS_STRIDE + sch * 8) = vb;
  }

  const v8f vzero = {};
  v8f c[2][4];
#pragma unroll
  for (int i = 0; i < 2; ++i)
#pragma unroll
    for (int j = 0; j < 4; ++j) c[i][j] = vzero;

  const int nk = K / BLK_K;
  for (int ik = 0; ik < nk; ++ik) {
    const int buf = ik & 1;
    __syncthreads();

    // early global loads of next tile into registers (overlap WMMAs below)
    const bool has_next = (ik + 1 < nk);
    const int k0n = (ik + 1) * BLK_K;
    v8h ga[2], gb[2];
    if (has_next) {
#pragma unroll
      for (int rep = 0; rep < 2; ++rep) {
        int row = srow0 + rep * 64;
        const f16* ap = A + (size_t)(m0 + row) * K + k0n + sch * 8;
        const f16* bp = Bm + (size_t)(n0 + row) * K + k0n + sch * 8;
        ga[rep] = *(const v8h*)ap;
        gb[rep] = *(const v8h*)bp;
        if (k0n + BLK_K < K) {  // global_prefetch_b8 one tile further
          __builtin_prefetch(ap + BLK_K, 0, 1);
          __builtin_prefetch(bp + BLK_K, 0, 1);
        }
      }
    }

    // all fragment ds_loads issued before any WMMA (staggered dscnt waits)
    v16h af[2], bf[4];
#pragma unroll
    for (int ms = 0; ms < 2; ++ms) {  // A frag: K chunks at 8h / 16+8h
      int r = wm * 32 + ms * 16 + ln;
      v8h lo = *(const v8h*)(sA[buf] + r * LDS_STRIDE + 8 * hh);
      v8h hi = *(const v8h*)(sA[buf] + r * LDS_STRIDE + 16 + 8 * hh);
      af[ms] = cat16(lo, hi);
    }
#pragma unroll
    for (int ns = 0; ns < 4; ++ns) {  // B frag: K chunks at 8h / 16+8h
      int col = wn * 64 + ns * 16 + ln;
      v8h lo = *(const v8h*)(sB[buf] + col * LDS_STRIDE + 8 * hh);
      v8h hi = *(const v8h*)(sB[buf] + col * LDS_STRIDE + 16 + 8 * hh);
      bf[ns] = cat16(lo, hi);
    }
#pragma unroll
    for (int ms = 0; ms < 2; ++ms)
#pragma unroll
      for (int ns = 0; ns < 4; ++ns) c[ms][ns] = wmma_f16(af[ms], bf[ns], c[ms][ns]);

    // deferred LDS stores of the staged tile (after this tile's ds_loads)
    if (has_next) {
#pragma unroll
      for (int rep = 0; rep < 2; ++rep) {
        int row = srow0 + rep * 64;
        *(v8h*)(sA[buf ^ 1] + row * LDS_STRIDE + sch * 8) = ga[rep];
        *(v8h*)(sB[buf ^ 1] + row * LDS_STRIDE + sch * 8) = gb[rep];
      }
    }
  }

  // epilogue: C frag element r at (m = r+8h, n = lane&15)
#pragma unroll
  for (int ms = 0; ms < 2; ++ms) {
#pragma unroll
    for (int ns = 0; ns < 4; ++ns) {
      if (MODE == MODE_VT) {
        // per-lane: 8 values along m = 8 consecutive s at fixed d -> one b128
        int n = n0 + wn * 64 + ns * 16 + ln;
        int head = n >> 7, d = n & 127;
        int mb = m0 + wm * 32 + ms * 16 + 8 * hh;  // + r, r=0..7
        int b = mb / S_, s = mb % S_;              // 8-aligned, no batch crossing
        v8h pk;
#pragma unroll
        for (int r = 0; r < 8; ++r) pk[r] = (f16)c[ms][ns][r];
        *(v8h*)((f16*)outp + (((size_t)(b * HEADS_ + head)) * HD_ + d) * S_ + s) = pk;
      } else {
#pragma unroll
        for (int r = 0; r < 8; ++r) {
          int m = m0 + wm * 32 + ms * 16 + r + 8 * hh;
          int n = n0 + wn * 64 + ns * 16 + ln;
          float v = c[ms][ns][r];
          if (MODE == MODE_F32) {
            ((float*)outp)[(size_t)m * Ncols + n] = v;
          } else if (MODE == MODE_F16) {
            ((f16*)outp)[(size_t)m * Ncols + n] = (f16)v;
          } else {  // MODE_ROT: partner channel (kd^1) in neighboring lane
            float pv = __shfl_xor(v, 1, 32);
            int b = m / S_, s = m % S_;
            int head = n >> 6, kd = n & 63;
            float cs = cosp[s * KD_ + kd];
            float sn = sinp[s * KD_ + kd];
            float res = (kd & 1) ? (v * cs + pv * sn) : (v * cs - pv * sn);
            res *= scale;
            ((f16*)outp)[(((size_t)(b * HEADS_ + head)) * S_ + s) * KD_ + kd] = (f16)res;
          }
        }
      }
    }
  }
}

// ---------------------------------------------------------------------------
// Retention core.  Per (b, head): 128 query rows/block (8 waves x 16 rows),
// streaming causal t-tiles of 32, double-buffered K/V in LDS.
// S^T = K*Q^T so the score C-frag converts IN-LANE into the PV A-frag.
// Decay mask, branchless + 1 exp2 per tile:
//   decay^(s-t) = decay^(s-t0) * decay^-(t-t0);  colpow precomputed per lane.
// acc = sum_t S[s,t] V[t,d]; rowsum = sum_t S[s,t];
// out = acc / max(|rowsum|,1) -> RMSNorm(HD) -> g*sigmoid(g) gate -> f16.
// ---------------------------------------------------------------------------
#define SK_STRIDE 72  // 32 x 64 halfs, padded
#define SV_STRIDE 40  // 128 x 32 halfs, padded

__global__ __launch_bounds__(256) void retention_kernel(
    const f16* __restrict__ qr, const f16* __restrict__ kr, const f16* __restrict__ vT,
    const f16* __restrict__ g, f16* __restrict__ gated) {
  __shared__ __align__(16) f16 sK[2][32 * SK_STRIDE];
  __shared__ __align__(16) f16 sV[2][128 * SV_STRIDE];

  const int b = blockIdx.z, head = blockIdx.y;
  const int s0 = blockIdx.x * 128;
  const int tid = threadIdx.x, wave = tid >> 5, lane = tid & 31;
  const int hh = lane >> 4, ln = lane & 15;
  const size_t bh = (size_t)(b * HEADS_ + head);
  const float log2d = log2f(1.f - exp2f(-(float)(5 + head)));
  const v8f vzero = {};

  // Q as B-fragments (col s = lane, K = kd): kd chunks at j*32 + 8h / j*32 + 16 + 8h
  v16h bq[2];
  const int srow = s0 + wave * 16 + ln;
  {
    const f16* qp = qr + (bh * S_ + srow) * KD_;
#pragma unroll
    for (int j = 0; j < 2; ++j) {
      v8h lo = *(const v8h*)(qp + j * 32 + 8 * hh);
      v8h hi = *(const v8h*)(qp + j * 32 + 16 + 8 * hh);
      bq[j] = cat16(lo, hi);
    }
  }

  // decay^-(t_local) for this lane's 16 fixed local t offsets (once)
  float colpow[2][8];
#pragma unroll
  for (int tstep = 0; tstep < 2; ++tstep)
#pragma unroll
    for (int r = 0; r < 8; ++r)
      colpow[tstep][r] = exp2f(-log2d * (float)(tstep * 16 + r + 8 * hh));

  // per-thread staging coordinates
  const int krow = tid >> 3, kch = tid & 7;  // K tile: 32 rows x 8 chunks
  const int vrow = tid >> 2, vch = tid & 3;  // V tile: rows +0/+64, 4 chunks

  // prologue: stage tile 0
  {
    v8h v = *(const v8h*)(kr + (bh * S_ + krow) * KD_ + kch * 8);
    *(v8h*)(sK[0] + krow * SK_STRIDE + kch * 8) = v;
#pragma unroll
    for (int rep = 0; rep < 2; ++rep) {
      int row = vrow + rep * 64;
      v8h w = *(const v8h*)(vT + (bh * HD_ + row) * S_ + vch * 8);
      *(v8h*)(sV[0] + row * SV_STRIDE + vch * 8) = w;
    }
  }

  v8f acc[8];
#pragma unroll
  for (int i = 0; i < 8; ++i) acc[i] = vzero;
  float rsum = 0.f;

  const int nt = (s0 + 128) / 32;
  for (int it = 0; it < nt; ++it) {
    const int buf = it & 1;
    const int t0 = it * 32;
    __syncthreads();

    // early global loads of next tile into registers
    const bool has_next = (it + 1 < nt);
    v8h gk, gv0, gv1;
    if (has_next) {
      const int t0n = t0 + 32;
      gk  = *(const v8h*)(kr + (bh * S_ + t0n + krow) * KD_ + kch * 8);
      gv0 = *(const v8h*)(vT + (bh * HD_ + vrow) * S_ + t0n + vch * 8);
      gv1 = *(const v8h*)(vT + (bh * HD_ + vrow + 64) * S_ + t0n + vch * 8);
    }

    // all fragment ds_loads up front: 8 (K) + 16 (V) in flight
    v16h ak[2][2];
#pragma unroll
    for (int tstep = 0; tstep < 2; ++tstep) {
      int trow = tstep * 16 + ln;
#pragma unroll
      for (int j = 0; j < 2; ++j) {
        v8h lo = *(const v8h*)(sK[buf] + trow * SK_STRIDE + j * 32 + 8 * hh);
        v8h hi = *(const v8h*)(sK[buf] + trow * SK_STRIDE + j * 32 + 16 + 8 * hh);
        ak[tstep][j] = cat16(lo, hi);
      }
    }
    v16h bv[8];
#pragma unroll
    for (int nd = 0; nd < 8; ++nd) {
      int d = nd * 16 + ln;
      v8h lo = *(const v8h*)(sV[buf] + d * SV_STRIDE + 8 * hh);
      v8h hi = *(const v8h*)(sV[buf] + d * SV_STRIDE + 16 + 8 * hh);
      bv[nd] = cat16(lo, hi);
    }

    // S^T tiles: D(16t x 16s) = Ktile(16t x 32kd) * Q^T(32kd x 16s)
    v8f cT[2];
#pragma unroll
    for (int tstep = 0; tstep < 2; ++tstep) {
      cT[tstep] = vzero;
#pragma unroll
      for (int j = 0; j < 2; ++j)
        cT[tstep] = wmma_f16(ak[tstep][j], bq[j], cT[tstep]);
    }

    // branchless decay mask (1 exp2/tile) + rowsum + in-lane cvt to A-frag:
    // A elem e: K=t_local = e+8h (e<8) / e+8+8h (e>=8)  ==  cT[e>>3][e&7]
    const float rowpow = exp2f(log2d * (float)(srow - t0));
    v16h as;
#pragma unroll
    for (int tstep = 0; tstep < 2; ++tstep) {
#pragma unroll
      for (int r = 0; r < 8; ++r) {
        int t = t0 + tstep * 16 + r + 8 * hh;
        float mv = (srow >= t) ? rowpow * colpow[tstep][r] : 0.f;  // v_cndmask
        float val = cT[tstep][r] * mv;
        rsum += val;
        as[tstep * 8 + r] = (f16)val;
      }
    }

    // PV: acc(16s x 16d) += S(16s x 32t) * V(32t x 16d)
#pragma unroll
    for (int nd = 0; nd < 8; ++nd) acc[nd] = wmma_f16(as, bv[nd], acc[nd]);

    // deferred LDS stores of the staged tile
    if (has_next) {
      *(v8h*)(sK[buf ^ 1] + krow * SK_STRIDE + kch * 8) = gk;
      *(v8h*)(sV[buf ^ 1] + vrow * SV_STRIDE + vch * 8) = gv0;
      *(v8h*)(sV[buf ^ 1] + (vrow + 64) * SV_STRIDE + vch * 8) = gv1;
    }
  }

  // full rowsum for column s: other 16 t-values live in the other half
  rsum += __shfl_xor(rsum, 16, 32);

  // redistribute per-row scalars: acc row m = r+8h needs rowsum of lane (r+8h)
  float inv[8];
#pragma unroll
  for (int r = 0; r < 8; ++r) {
    float rs = __shfl(rsum, r + 8 * hh, 32);
    float dn = fabsf(rs);
    inv[r] = 1.f / (dn < 1.f ? 1.f : dn);
  }
#pragma unroll
  for (int nd = 0; nd < 8; ++nd)
#pragma unroll
    for (int r = 0; r < 8; ++r) acc[nd][r] *= inv[r];

  // RMS over HD=128: per-lane partial, reduce across the 16 lanes of the half
  float ssq[8];
#pragma unroll
  for (int r = 0; r < 8; ++r) {
    ssq[r] = 0.f;
#pragma unroll
    for (int nd = 0; nd < 8; ++nd) { float v = acc[nd][r]; ssq[r] += v * v; }
  }
#pragma unroll
  for (int r = 0; r < 8; ++r) {
#pragma unroll
    for (int off = 1; off < 16; off <<= 1) ssq[r] += __shfl_xor(ssq[r], off, 32);
  }
  float rn[8];
#pragma unroll
  for (int r = 0; r < 8; ++r) rn[r] = rsqrtf(ssq[r] * (1.f / 128.f) + 1e-5f);

  // SiLU gate and f16 store (row-major A-operand for the final GEMM)
#pragma unroll
  for (int nd = 0; nd < 8; ++nd) {
#pragma unroll
    for (int r = 0; r < 8; ++r) {
      int m = s0 + wave * 16 + r + 8 * hh;
      int col = head * HD_ + nd * 16 + ln;
      float gv = (float)g[((size_t)b * S_ + m) * VDIM_ + col];
      float sg = 1.f / (1.f + __expf(-gv));
      float val = acc[nd][r] * rn[r] * gv * sg;
      gated[((size_t)b * S_ + m) * VDIM_ + col] = (f16)val;
    }
  }
}

// ---------------------------------------------------------------------------
// launch
// ---------------------------------------------------------------------------
extern "C" void kernel_launch(void* const* d_in, const int* in_sizes, int n_in,
                              void* d_out, int out_size, void* d_ws, size_t ws_size,
                              hipStream_t stream) {
  (void)in_sizes; (void)n_in; (void)out_size; (void)ws_size;
  const float* x    = (const float*)d_in[0];
  const float* sinp = (const float*)d_in[1];
  const float* cosp = (const float*)d_in[2];
  // d_in[3] = mask: unused (decay computed analytically in-kernel)
  const float* Wq   = (const float*)d_in[4];
  const float* Wk   = (const float*)d_in[5];
  const float* Wv   = (const float*)d_in[6];
  const float* Wg   = (const float*)d_in[7];
  const float* Wo   = (const float*)d_in[8];
  float* out = (float*)d_out;

  char* ws = (char*)d_ws;
  size_t off = 0;
  auto alloc = [&](size_t bytes) -> char* {
    char* p = ws + off;
    off += (bytes + 255) & ~(size_t)255;
    return p;
  };
  const size_t NT = (size_t)B_ * S_;                  // 4096 tokens
  f16* xh    = (f16*)alloc(NT * H_ * 2);              // 8 MB
  f16* Wq16  = (f16*)alloc((size_t)H_ * H_ * 2);      // 2 MB
  f16* Wk16  = (f16*)alloc((size_t)H_ * H_ * 2);      // 2 MB
  f16* Wv16  = (f16*)alloc((size_t)VDIM_ * H_ * 2);   // 4 MB
  f16* Wg16  = (f16*)alloc((size_t)VDIM_ * H_ * 2);   // 4 MB
  f16* Wo16  = (f16*)alloc((size_t)H_ * VDIM_ * 2);   // 4 MB
  f16* qrbuf = (f16*)alloc(NT * H_ * 2);              // 8 MB  [b,h,s,kd]
  f16* krbuf = (f16*)alloc(NT * H_ * 2);              // 8 MB  [b,h,s,kd]
  f16* vTbuf = (f16*)alloc(NT * VDIM_ * 2);           // 16 MB [b,h,d,s]
  f16* gbuf  = (f16*)alloc(NT * VDIM_ * 2);           // 16 MB [b,s,v]
  f16* gated = (f16*)alloc(NT * VDIM_ * 2);           // 16 MB [b,s,v]

  auto cvt = [&](const float* s, f16* d, int n) {
    cvt_f32_f16<<<(n + 255) / 256, 256, 0, stream>>>(s, d, n);
  };
  cvt(x,  xh,   (int)(NT * H_));
  cvt(Wq, Wq16, H_ * H_);
  cvt(Wk, Wk16, H_ * H_);
  cvt(Wv, Wv16, VDIM_ * H_);
  cvt(Wg, Wg16, VDIM_ * H_);
  cvt(Wo, Wo16, H_ * VDIM_);

  const float kscale = 0.125f;  // KD^-0.5
  dim3 blk(256);
  gemm_wmma<MODE_ROT><<<dim3(NT / BLK_M, H_ / BLK_N), blk, 0, stream>>>(
      xh, Wq16, qrbuf, H_, H_, sinp, cosp, 1.0f);
  gemm_wmma<MODE_ROT><<<dim3(NT / BLK_M, H_ / BLK_N), blk, 0, stream>>>(
      xh, Wk16, krbuf, H_, H_, sinp, cosp, kscale);
  gemm_wmma<MODE_VT><<<dim3(NT / BLK_M, VDIM_ / BLK_N), blk, 0, stream>>>(
      xh, Wv16, vTbuf, H_, VDIM_, nullptr, nullptr, 1.0f);
  gemm_wmma<MODE_F16><<<dim3(NT / BLK_M, VDIM_ / BLK_N), blk, 0, stream>>>(
      xh, Wg16, gbuf, H_, VDIM_, nullptr, nullptr, 1.0f);

  retention_kernel<<<dim3(S_ / 128, HEADS_, B_), blk, 0, stream>>>(
      qrbuf, krbuf, vTbuf, gbuf, gated);

  gemm_wmma<MODE_F32><<<dim3(NT / BLK_M, H_ / BLK_N), blk, 0, stream>>>(
      gated, Wo16, out, VDIM_, H_, nullptr, nullptr, 1.0f);
}
